// NerfMLP_32624571580760
// MI455X (gfx1250) — compile-verified, matrix operand order checked by reading, not run on hardware
//
#include <hip/hip_runtime.h>
#include <hip/hip_bf16.h>

typedef __attribute__((ext_vector_type(16))) _Float16 v16h;
typedef __attribute__((ext_vector_type(8)))  _Float16 v8h;
typedef __attribute__((ext_vector_type(8)))  float    v8f;

#define S_SAMPLES 64
#define B_RAYS    4096
#define NPTS      (S_SAMPLES * B_RAYS)
#define MT        128      // points (rows) per workgroup
#define HS        296      // f16 per h-row: 288 used + pad -> conflict-free b128 row loads

// ---------------- weight transpose/convert: w[K][N] f32 -> wT[Npad][Kpad] f16 ----
__global__ void transpose_w_kernel(const float* __restrict__ src, _Float16* __restrict__ dst,
                                   int K, int N, int Kpad, int Npad) {
    int idx = blockIdx.x * 256 + threadIdx.x;
    int total = Kpad * Npad;
    if (idx >= total) return;
    int n = idx / Kpad;
    int k = idx - n * Kpad;
    float v = (k < K && n < N) ? src[k * N + n] : 0.0f;
    dst[idx] = (_Float16)v;
}

// ---------------- CDNA5 async global->LDS copy (ASYNCcnt-tracked) -----------------
__device__ __forceinline__ void async_cp16(const _Float16* g, unsigned lds_addr) {
    asm volatile("global_load_async_to_lds_b128 %0, %1, off"
                 :: "v"(lds_addr), "v"((unsigned long long)(uintptr_t)g)
                 : "memory");
}

#if defined(__has_builtin)
#if __has_builtin(__builtin_amdgcn_s_wait_asynccnt)
#define WAIT_ASYNC() __builtin_amdgcn_s_wait_asynccnt(0)
#else
#define WAIT_ASYNC() asm volatile("s_wait_asynccnt 0x0" ::: "memory")
#endif
#else
#define WAIT_ASYNC() asm volatile("s_wait_asynccnt 0x0" ::: "memory")
#endif

// stage wT[:,koffw:koffw+KC] (row stride kpadw) into LDS dst[npad][KC], async
template<int KC>
__device__ __forceinline__ void stage_async(const _Float16* __restrict__ wT, int kpadw, int koffw,
                                            int npad, _Float16* dst, int tid) {
    const int cpr = KC >> 3;                   // 16B chunks per row
    const int total = npad * cpr;
    const unsigned dbase = (unsigned)(uintptr_t)dst;   // LDS offset = low 32 bits of flat addr
    for (int c = tid; c < total; c += 256) {
        int row = c / cpr;
        int ch  = c - row * cpr;
        async_cp16(wT + row * kpadw + koffw + ch * 8,
                   dbase + (unsigned)((row * KC + ch * 8) * 2));
    }
}

// ---------------- fragment load: two contiguous 16B chunks -> v16h -----------------
__device__ __forceinline__ v16h frag16(const _Float16* p) {
    v8h a0 = *(const v8h*)(p);
    v8h a1 = *(const v8h*)(p + 16);
    return __builtin_shufflevector(a0, a1, 0,1,2,3,4,5,6,7,8,9,10,11,12,13,14,15);
}

// acc[nt] += A(h rows mb*16.., cols koff..) x B(wbuf n-tile nt); B stride = KSTEPS*32
template<int NT, int KSTEPS>
__device__ __forceinline__ void wmma_block(const _Float16* hsm, int koff,
                                           const _Float16* wsm, v8f* acc,
                                           int mb, int lane) {
    const int lane16 = lane & 15;
    const int lanehi = lane >> 4;
    const int KC = KSTEPS * 32;
    const _Float16* abase = hsm + (mb * 16 + lane16) * HS + koff + lanehi * 8;
#pragma unroll
    for (int k = 0; k < KSTEPS; ++k) {
        v16h a = frag16(abase + k * 32);
#pragma unroll
        for (int nt = 0; nt < NT; ++nt) {
            const _Float16* bp = wsm + (nt * 16 + lane16) * KC + k * 32 + lanehi * 8;
            v16h b = frag16(bp);
            acc[nt] = __builtin_amdgcn_wmma_f32_16x16x32_f16(
                false, a, false, b, (short)0, acc[nt], false, false);
        }
    }
}

__device__ __forceinline__ void zero_acc(v8f* acc, int n) {
    for (int nt = 0; nt < n; ++nt) {
#pragma unroll
        for (int e = 0; e < 8; ++e) acc[nt][e] = 0.0f;
    }
}

// bias + relu + f16 store back into h cols 0..255
__device__ __forceinline__ void store_relu(_Float16* hsm, const v8f* acc,
                                           const float* __restrict__ bias,
                                           int mb, int lane) {
    const int lane16 = lane & 15;
    const int lanehi = lane >> 4;
#pragma unroll
    for (int nt = 0; nt < 16; ++nt) {
        float bn = bias[nt * 16 + lane16];
#pragma unroll
        for (int v = 0; v < 8; ++v) {
            float r = acc[nt][v] + bn;
            r = r > 0.0f ? r : 0.0f;
            hsm[(mb * 16 + lanehi * 8 + v) * HS + nt * 16 + lane16] = (_Float16)r;
        }
    }
}

// ---------------------------------- main MLP kernel --------------------------------
__global__ void __launch_bounds__(256, 1)
nerf_mlp_kernel(const float* __restrict__ x, const float* __restrict__ offsets,
                const _Float16* wT0, const _Float16* wT1, const _Float16* wT2,
                const _Float16* wT3, const _Float16* wT4, const _Float16* wT5,
                const _Float16* wT6, const _Float16* wT7,
                const float* b0, const float* b1, const float* b2, const float* b3,
                const float* b4, const float* b5, const float* b6, const float* b7,
                const float* __restrict__ w8, const float* __restrict__ b8,
                float4* __restrict__ colden) {
    __shared__ _Float16 h[MT * HS];          // activations; enc kept at cols 256..287
    __shared__ _Float16 wbuf0[256 * 64];     // double-buffered weight K-chunks (32KB each)
    __shared__ _Float16 wbuf1[256 * 64];
    __shared__ float    dens[MT];

    const int tid  = threadIdx.x;
    const int lane = tid & 31;
    const int mb   = tid >> 5;               // wave id = M-block (16 rows each)

    // ---- phase 0: positional encoding into h[:,256..287] ----
    if (tid < MT) {
        int p = blockIdx.x * MT + tid;
        int s = p >> 12;                     // B=4096
        int b = p & (B_RAYS - 1);
        float off = offsets[s * B_RAYS + b];
        float z = 2.0f + (float)s * 0.0625f + off * 0.0625f;   // NEAR + (s+off)*(FAR-NEAR)/S
        float ox = x[b*6+0], oy = x[b*6+1], oz = x[b*6+2];
        float dx = x[b*6+3], dy = x[b*6+4], dz = x[b*6+5];
        float pos[3] = { ox + dx*z, oy + dy*z, oz + dz*z };
        const float PI = 3.14159265358979323846f;
        _Float16* e = h + tid * HS + 256;
#pragma unroll
        for (int j = 0; j < 3; ++j) {
            float f = 0.25f * PI;
#pragma unroll
            for (int l = 0; l < 5; ++l) {
                float ang = pos[j] * f;
                e[j*10 + l*2 + 0] = (_Float16)__sinf(ang);
                e[j*10 + l*2 + 1] = (_Float16)__cosf(ang);
                f *= 2.0f;
            }
        }
        e[30] = (_Float16)0.0f;
        e[31] = (_Float16)0.0f;
    }

    v8f acc[16];
    _Float16* wb[2] = { wbuf0, wbuf1 };
    int cur = 0;

    // prefetch into wb[cur^1]
#define PF(WT, KPW, KOFF, NPAD, KC) stage_async<KC>(WT, KPW, KOFF, NPAD, wb[cur ^ 1], tid)
#define NOPF ((void)0)
    // consume one staged chunk; issue next-stage prefetch while computing
#define STEP(NT, KS, KOFFH, PRE)                                    \
    do {                                                            \
        WAIT_ASYNC();                                               \
        __syncthreads();                                            \
        PRE;                                                        \
        wmma_block<NT, KS>(h, KOFFH, wb[cur], acc, mb, lane);       \
        cur ^= 1;                                                   \
    } while (0)

    // ---- prologue: stage layer0 weights (K=32) ----
    stage_async<32>(wT0, 32, 0, 256, wb[cur], tid);

    // ---- layer 0: K=32 (enc @ h cols 256..287) ----
    zero_acc(acc, 16);
    STEP(16, 1, 256, PF(wT1, 256, 0, 256, 64));
    __syncthreads();
    store_relu(h, acc, b0, mb, lane);

    // ---- layer 1..3: K=256 in four 64-chunks, async double-buffered ----
    zero_acc(acc, 16);
    STEP(16, 2, 0,   PF(wT1, 256, 64,  256, 64));
    STEP(16, 2, 64,  PF(wT1, 256, 128, 256, 64));
    STEP(16, 2, 128, PF(wT1, 256, 192, 256, 64));
    STEP(16, 2, 192, PF(wT2, 256, 0,   256, 64));
    __syncthreads();
    store_relu(h, acc, b1, mb, lane);

    zero_acc(acc, 16);
    STEP(16, 2, 0,   PF(wT2, 256, 64,  256, 64));
    STEP(16, 2, 64,  PF(wT2, 256, 128, 256, 64));
    STEP(16, 2, 128, PF(wT2, 256, 192, 256, 64));
    STEP(16, 2, 192, PF(wT3, 256, 0,   256, 64));
    __syncthreads();
    store_relu(h, acc, b2, mb, lane);

    zero_acc(acc, 16);
    STEP(16, 2, 0,   PF(wT3, 256, 64,  256, 64));
    STEP(16, 2, 64,  PF(wT3, 256, 128, 256, 64));
    STEP(16, 2, 128, PF(wT3, 256, 192, 256, 64));
    STEP(16, 2, 192, PF(wT4, 288, 0,   256, 64));
    __syncthreads();
    store_relu(h, acc, b3, mb, lane);

    // ---- layer 4: K=288 = [h(256) | enc(30+pad)] ----
    zero_acc(acc, 16);
    STEP(16, 2, 0,   PF(wT4, 288, 64,  256, 64));
    STEP(16, 2, 64,  PF(wT4, 288, 128, 256, 64));
    STEP(16, 2, 128, PF(wT4, 288, 192, 256, 64));
    STEP(16, 2, 192, PF(wT4, 288, 256, 256, 32));
    STEP(16, 1, 256, PF(wT5, 256, 0,   256, 64));
    __syncthreads();
    store_relu(h, acc, b4, mb, lane);

    // ---- layers 5..6 ----
    zero_acc(acc, 16);
    STEP(16, 2, 0,   PF(wT5, 256, 64,  256, 64));
    STEP(16, 2, 64,  PF(wT5, 256, 128, 256, 64));
    STEP(16, 2, 128, PF(wT5, 256, 192, 256, 64));
    STEP(16, 2, 192, PF(wT6, 256, 0,   256, 64));
    __syncthreads();
    store_relu(h, acc, b5, mb, lane);

    zero_acc(acc, 16);
    STEP(16, 2, 0,   PF(wT6, 256, 64,  256, 64));
    STEP(16, 2, 64,  PF(wT6, 256, 128, 256, 64));
    STEP(16, 2, 128, PF(wT6, 256, 192, 256, 64));
    STEP(16, 2, 192, PF(wT7, 256, 0,   144, 64));
    __syncthreads();
    store_relu(h, acc, b6, mb, lane);

    // ---- layer 7: K=256, N=129 (padded to 144 -> 9 n-tiles) ----
    zero_acc(acc, 9);
    STEP(9, 2, 0,    PF(wT7, 256, 64,  144, 64));
    STEP(9, 2, 64,   PF(wT7, 256, 128, 144, 64));
    STEP(9, 2, 128,  PF(wT7, 256, 192, 144, 64));
    STEP(9, 2, 192,  NOPF);
    __syncthreads();
    {
        const int lane16 = lane & 15;
        const int lanehi = lane >> 4;
#pragma unroll
        for (int nt = 0; nt < 9; ++nt) {
            int n = nt * 16 + lane16;
            float bn = (n < 129) ? b7[n] : 0.0f;
#pragma unroll
            for (int v = 0; v < 8; ++v) {
                int row = mb * 16 + lanehi * 8 + v;
                float r = acc[nt][v] + bn;
                if (n == 0) {
                    dens[row] = r;                       // raw density
                } else if (n < 129) {
                    float f = r > 0.0f ? r : 0.0f;       // relu(feat)
                    h[row * HS + (n - 1)] = (_Float16)f; // feat -> cols 0..127
                }
            }
        }
    }
    __syncthreads();

    // ---- layer 8 (131 -> 3) + sigmoid, pack rgb + density ----
    if (tid < MT) {
        int r = tid;
        int p = blockIdx.x * MT + r;
        int b = p & (B_RAYS - 1);
        float dx = x[b*6+3], dy = x[b*6+4], dz = x[b*6+5];
        float inorm = rsqrtf(dx*dx + dy*dy + dz*dz);
        float vd[3] = { dx * inorm, dy * inorm, dz * inorm };
        float c0 = b8[0], c1 = b8[1], c2 = b8[2];
        for (int k = 0; k < 128; ++k) {
            float f = (float)h[r * HS + k];
            c0 += f * w8[k*3 + 0];
            c1 += f * w8[k*3 + 1];
            c2 += f * w8[k*3 + 2];
        }
#pragma unroll
        for (int k = 0; k < 3; ++k) {
            c0 += vd[k] * w8[(128 + k)*3 + 0];
            c1 += vd[k] * w8[(128 + k)*3 + 1];
            c2 += vd[k] * w8[(128 + k)*3 + 2];
        }
        float4 o;
        o.x = 1.0f / (1.0f + __expf(-c0));
        o.y = 1.0f / (1.0f + __expf(-c1));
        o.z = 1.0f / (1.0f + __expf(-c2));
        o.w = dens[r];
        colden[p] = o;
    }
}

// ------------------------------ volume rendering ------------------------------------
__global__ void composite_kernel(const float4* __restrict__ colden,
                                 const float* __restrict__ x,
                                 const float* __restrict__ offsets,
                                 float* __restrict__ out) {
    int b = blockIdx.x * blockDim.x + threadIdx.x;
    if (b >= B_RAYS) return;
    float dx = x[b*6+3], dy = x[b*6+4], dz = x[b*6+5];
    float norm = sqrtf(dx*dx + dy*dy + dz*dz);
    float T = 1.0f, r = 0.0f, g = 0.0f, bl = 0.0f;
    float zn_cur = (2.0f + offsets[b] * 0.0625f) * norm;
    for (int s = 0; s < S_SAMPLES; ++s) {
        float dist;
        if (s < S_SAMPLES - 1) {
            float zn_next = (2.0f + ((float)(s + 1) + offsets[(s + 1) * B_RAYS + b]) * 0.0625f) * norm;
            dist = zn_next - zn_cur;
            zn_cur = zn_next;
        } else {
            dist = 1e10f;
        }
        float4 cd = colden[s * B_RAYS + b];
        float sigma = cd.w > 0.0f ? cd.w : 0.0f;
        float alpha = 1.0f - __expf(-sigma * dist);
        T *= (1.0f - alpha + 1e-10f);
        float w = alpha * T;          // inclusive cumprod per reference
        r  += cd.x * w;
        g  += cd.y * w;
        bl += cd.z * w;
    }
    out[b*3 + 0] = r;
    out[b*3 + 1] = g;
    out[b*3 + 2] = bl;
}

// ------------------------------------ launch ----------------------------------------
extern "C" void kernel_launch(void* const* d_in, const int* in_sizes, int n_in,
                              void* d_out, int out_size, void* d_ws, size_t ws_size,
                              hipStream_t stream) {
    const float* x       = (const float*)d_in[0];
    const float* offsets = (const float*)d_in[1];
    const float* w[8];
    const float* bias[8];
    for (int i = 0; i < 8; ++i) {
        w[i]    = (const float*)d_in[2 + 2 * i];
        bias[i] = (const float*)d_in[3 + 2 * i];
    }
    const float* w8 = (const float*)d_in[18];
    const float* b8 = (const float*)d_in[19];

    const int Ks[8]   = {30, 256, 256, 256, 286, 256, 256, 256};
    const int Ns[8]   = {256, 256, 256, 256, 256, 256, 256, 129};
    const int Kpad[8] = {32, 256, 256, 256, 288, 256, 256, 256};
    const int Npad[8] = {256, 256, 256, 256, 256, 256, 256, 144};

    _Float16* wsbase = (_Float16*)d_ws;
    _Float16* wT[8];
    size_t off = 0;
    for (int i = 0; i < 8; ++i) {
        wT[i] = wsbase + off;
        off += (size_t)Kpad[i] * Npad[i];
    }
    size_t bytes = off * sizeof(_Float16);
    bytes = (bytes + 255) & ~(size_t)255;
    float4* colden = (float4*)((char*)d_ws + bytes);

    for (int i = 0; i < 8; ++i) {
        int total = Kpad[i] * Npad[i];
        transpose_w_kernel<<<(total + 255) / 256, 256, 0, stream>>>(
            w[i], wT[i], Ks[i], Ns[i], Kpad[i], Npad[i]);
    }

    nerf_mlp_kernel<<<NPTS / MT, 256, 0, stream>>>(
        x, offsets,
        wT[0], wT[1], wT[2], wT[3], wT[4], wT[5], wT[6], wT[7],
        bias[0], bias[1], bias[2], bias[3], bias[4], bias[5], bias[6], bias[7],
        w8, b8, colden);

    composite_kernel<<<B_RAYS / 256, 256, 0, stream>>>(colden, x, offsets, (float*)d_out);
}
